// TransformerSpeculationModel_33621003993540
// MI455X (gfx1250) — compile-verified, matrix operand order checked by reading, not run on hardware
//
#include <hip/hip_runtime.h>
#include <hip/hip_bf16.h>

// ---------------------------------------------------------------------------
// TransformerSpeculationModel on MI455X (gfx1250, wave32, WMMA).
// One workgroup per batch element; whole 32x512 residual stream lives in LDS
// across all 6 layers. All GEMMs via v_wmma_f32_16x16x32_f16.
// B-tiles (weights / K / V) are loaded once and shared across both M-tiles
// via dual accumulators -> halves L2 weight traffic.
// ---------------------------------------------------------------------------

#define B_   4096
#define S_   32
#define H_   512
#define NH_  8
#define HD_  64
#define NL_  6
#define E_   128
#define LN_EPS 1e-5f

typedef _Float16 f16;
typedef __attribute__((ext_vector_type(16))) _Float16 v16h;
typedef __attribute__((ext_vector_type(8)))  _Float16 v8h;
typedef __attribute__((ext_vector_type(8)))  float    v8f;

// ---- f32 -> f16 weight conversion (runs once; weights then live in L2) ----
__global__ void cvt_f32_f16(const float* __restrict__ src, f16* __restrict__ dst, int n) {
    for (int i = blockIdx.x * blockDim.x + threadIdx.x; i < n; i += gridDim.x * blockDim.x)
        dst[i] = (f16)src[i];
}

// ---- WMMA tile loaders -----------------------------------------------------
// A-matrix (or B from row-major W so that B[k][n] = W[n][k]):
// lane<16: row = row0+lane, K = k0+{0..7, 16..23}; lane>=16: K = k0+{8..15, 24..31}.
// Requires (base,ld,k0) to be multiples of 8 halves -> 16B-aligned vector loads.
__device__ __forceinline__ v16h load_tile_rowmajor(const f16* base, int ld, int row0, int k0, int lane) {
    int r  = lane & 15;
    int kb = (lane < 16) ? 0 : 8;
    const f16* p = base + (size_t)(row0 + r) * ld + k0 + kb;
    union { v16h v; v8h h[2]; } u;
    u.h[0] = *(const v8h*)(p);
    u.h[1] = *(const v8h*)(p + 16);
    return u.v;
}

// B tile where the source is row-major along K (B[k][n] = src[k][n]) -> gather.
__device__ __forceinline__ v16h load_tile_colgather(const f16* base, int ld, int k0, int n0, int lane) {
    int n  = n0 + (lane & 15);
    int kb = (lane < 16) ? 0 : 8;
    v16h v;
#pragma unroll
    for (int e = 0; e < 8; ++e) {
        v[e]     = base[(size_t)(k0 + kb + e) * ld + n];
        v[e + 8] = base[(size_t)(k0 + 16 + kb + e) * ld + n];
    }
    return v;
}

// ---- Tiled GEMM:  out[32 x N] = A[32 x K] @ W[N x K]^T + bias --------------
// Dual accumulators: each weight B-tile is loaded once and used for both
// 16-row M-tiles (M = 32).
// ACC=false: write f16 (optionally ReLU) to out16 (ld=ldo)
// ACC=true : accumulate f32 into outf (residual add), ld=ldof
template <bool ACC, bool RELU>
__device__ __forceinline__ void gemm_x_wT(const f16* A, int lda,
                                          const f16* W, int K,
                                          const float* __restrict__ bias,
                                          f16* out16, int ldo,
                                          float* outf, int ldof,
                                          int Ntiles, int wave, int lane) {
    for (int nt = wave; nt < Ntiles; nt += 8) {
        const int n0 = nt * 16;
        // prefetch the next weight tile this wave will touch (global_prefetch_b8)
        if (nt + 8 < Ntiles)
            __builtin_prefetch(W + (size_t)(n0 + 128) * K, 0, 1);
        v8f acc0 = {};
        v8f acc1 = {};
        for (int k0 = 0; k0 < K; k0 += 32) {
            v16h b  = load_tile_rowmajor(W, K,  n0, k0, lane);
            v16h a0 = load_tile_rowmajor(A, lda, 0,  k0, lane);
            v16h a1 = load_tile_rowmajor(A, lda, 16, k0, lane);
            acc0 = __builtin_amdgcn_wmma_f32_16x16x32_f16(
                false, a0, false, b, (short)0, acc0, false, false);
            acc1 = __builtin_amdgcn_wmma_f32_16x16x32_f16(
                false, a1, false, b, (short)0, acc1, false, false);
        }
        const int n  = n0 + (lane & 15);
        const int mh = (lane & 16) ? 8 : 0;
        const float bv = bias ? bias[n] : 0.0f;
#pragma unroll
        for (int r = 0; r < 8; ++r) {
            float v0 = acc0[r] + bv;
            float v1 = acc1[r] + bv;
            if (ACC) {
                outf[(size_t)(mh + r) * ldof + n]      += v0;
                outf[(size_t)(16 + mh + r) * ldof + n] += v1;
            } else {
                if (RELU) { v0 = v0 > 0.0f ? v0 : 0.0f; v1 = v1 > 0.0f ? v1 : 0.0f; }
                out16[(size_t)(mh + r) * ldo + n]      = (f16)v0;
                out16[(size_t)(16 + mh + r) * ldo + n] = (f16)v1;
            }
        }
    }
}

// ---- LayerNorm over H=512, 8 threads per row (wave32 shfl butterflies) -----
__device__ __forceinline__ void layernorm_rows(float* x, f16* stage,
                                               const float* __restrict__ w,
                                               const float* __restrict__ b, int tid) {
    const int row = tid >> 3;   // 0..31
    const int sub = tid & 7;    // 0..7, 64 elems each
    float* xr = x + (size_t)row * H_ + sub * 64;
    float s = 0.0f, s2 = 0.0f;
#pragma unroll 8
    for (int j = 0; j < 64; ++j) { float v = xr[j]; s += v; s2 += v * v; }
#pragma unroll
    for (int off = 1; off < 8; off <<= 1) {
        s  += __shfl_xor(s,  off, 8);
        s2 += __shfl_xor(s2, off, 8);
    }
    const float mu  = s * (1.0f / H_);
    const float var = s2 * (1.0f / H_) - mu * mu;
    const float rs  = rsqrtf(var + LN_EPS);
#pragma unroll 8
    for (int j = 0; j < 64; ++j) {
        int k = sub * 64 + j;
        float v = (xr[j] - mu) * rs * w[k] + b[k];
        xr[j] = v;
        if (stage) stage[(size_t)row * H_ + k] = (f16)v;
    }
}

// ---- Fused whole-model kernel: one block (8 wave32) per batch element ------
__global__ __launch_bounds__(256)
void fused_transformer(const int*   __restrict__ expert_sequence,  // (B,S)
                       const int*   __restrict__ layer_id,         // (B,)
                       const float* __restrict__ expert_table,     // (E,H)
                       const float* __restrict__ layer_table,      // (12,H)
                       const float* __restrict__ pos_table,        // (CTX,H)
                       const f16*   __restrict__ wqkv,             // (NL,3H,H) f16
                       const float* __restrict__ qkv_b,
                       const f16*   __restrict__ wout,             // (NL,H,H) f16
                       const float* __restrict__ out_b,
                       const float* __restrict__ ln1_w, const float* __restrict__ ln1_b,
                       const f16*   __restrict__ wff1,             // (NL,2H,H) f16
                       const float* __restrict__ ff1_b,
                       const f16*   __restrict__ wff2,             // (NL,H,2H) f16
                       const float* __restrict__ ff2_b,
                       const float* __restrict__ ln2_w, const float* __restrict__ ln2_b,
                       const float* __restrict__ head_w,           // (NL,E,H) f32
                       const float* __restrict__ head_b,
                       const float* __restrict__ conf_w,           // (1,H)
                       const float* __restrict__ conf_b,
                       float* __restrict__ out_pred,               // (B,E)
                       float* __restrict__ out_conf)               // (B,)
{
    extern __shared__ char smem[];
    float* xf32 = (float*)smem;                      // 32x512 f32   (64 KB)
    f16*   A16  = (f16*)(smem + 65536);              // 32x1024 f16  (64 KB) staging
    f16*   O16  = (f16*)(smem + 131072);             // 32x1536 f16  (96 KB) qkv / ff-hidden
    float* scoresF = (float*)A16;                    // 8 heads x 32x32 f32 (32 KB, aliases A16 lo)
    f16*   att16   = A16 + 16384;                    // 8 heads x 32x32 f16 (16 KB, aliases A16 hi)

    const int b    = blockIdx.x;
    const int tid  = threadIdx.x;
    const int lane = tid & 31;
    const int wave = tid >> 5;
    const float scale = 0.125f;                      // 1/sqrt(64)

    // ---- Embedding: x = expert + layer + pos (S == CTX so s % CTX == s) ----
    const int lid = layer_id[b];
    for (int idx = tid; idx < S_ * H_; idx += 256) {
        int s = idx >> 9, k = idx & (H_ - 1);
        int e = expert_sequence[b * S_ + s];
        xf32[idx] = expert_table[(size_t)e * H_ + k]
                  + layer_table[(size_t)lid * H_ + k]
                  + pos_table[(size_t)s * H_ + k];
    }
    __syncthreads();

    for (int i = 0; i < NL_; ++i) {
        const f16*   Wq = wqkv + (size_t)i * 3 * H_ * H_;
        const float* bq = qkv_b + (size_t)i * 3 * H_;
        const f16*   Wo = wout + (size_t)i * H_ * H_;
        const float* bo = out_b + (size_t)i * H_;
        const f16*   W1 = wff1 + (size_t)i * 2 * H_ * H_;
        const float* b1 = ff1_b + (size_t)i * 2 * H_;
        const f16*   W2 = wff2 + (size_t)i * H_ * 2 * H_;
        const float* b2 = ff2_b + (size_t)i * H_;

        // (a) stage x -> f16
        for (int idx = tid; idx < S_ * H_; idx += 256) A16[idx] = (f16)xf32[idx];
        __syncthreads();

        // (b) QKV: O16[32 x 1536] = x @ Wq^T + bq
        gemm_x_wT<false, false>(A16, H_, Wq, H_, bq, O16, 3 * H_, nullptr, 0, 96, wave, lane);
        __syncthreads();

        // (c) attention scores per head (wave h): S = q @ k^T  (M=N=32, K=64)
        {
            const int h = wave;
            const f16* qb = O16 + h * HD_;
            const f16* kb = O16 + H_ + h * HD_;
            float* sc = scoresF + h * S_ * S_;
#pragma unroll
            for (int nt = 0; nt < 2; ++nt) {
                v8f acc0 = {};
                v8f acc1 = {};
#pragma unroll
                for (int kt = 0; kt < 2; ++kt) {
                    v16h bm = load_tile_rowmajor(kb, 3 * H_, nt * 16, kt * 32, lane);
                    v16h a0 = load_tile_rowmajor(qb, 3 * H_, 0,  kt * 32, lane);
                    v16h a1 = load_tile_rowmajor(qb, 3 * H_, 16, kt * 32, lane);
                    acc0 = __builtin_amdgcn_wmma_f32_16x16x32_f16(
                        false, a0, false, bm, (short)0, acc0, false, false);
                    acc1 = __builtin_amdgcn_wmma_f32_16x16x32_f16(
                        false, a1, false, bm, (short)0, acc1, false, false);
                }
                int n  = nt * 16 + (lane & 15);
                int mh = (lane & 16) ? 8 : 0;
#pragma unroll
                for (int r = 0; r < 8; ++r) {
                    sc[(mh + r) * S_ + n]      = acc0[r];
                    sc[(16 + mh + r) * S_ + n] = acc1[r];
                }
            }
        }
        __syncthreads();

        // (d) softmax: one thread per (head,row) -> 8*32 = 256 threads
        {
            const int h = tid >> 5, row = tid & 31;
            float* sc = scoresF + h * S_ * S_ + row * S_;
            float mx = -3.0e38f;
#pragma unroll
            for (int j = 0; j < S_; ++j) mx = fmaxf(mx, sc[j]);
            float sum = 0.0f, e[S_];
#pragma unroll
            for (int j = 0; j < S_; ++j) { e[j] = __expf((sc[j] - mx) * scale); sum += e[j]; }
            float inv = 1.0f / sum;
            f16* at = att16 + h * S_ * S_ + row * S_;
#pragma unroll
            for (int j = 0; j < S_; ++j) at[j] = (f16)(e[j] * inv);
        }
        __syncthreads();

        // (e) o = att @ v per head (M=32, N=64, K=32); o -> A16[32 x 512]
        {
            const int h = wave;
            const f16* at = att16 + h * S_ * S_;        // ld = 32
            const f16* vb = O16 + 2 * H_ + h * HD_;     // row-major [k][n], ld = 1536
#pragma unroll
            for (int nt = 0; nt < 4; ++nt) {
                v16h bm = load_tile_colgather(vb, 3 * H_, 0, nt * 16, lane);
                v16h a0 = load_tile_rowmajor(at, S_, 0,  0, lane);
                v16h a1 = load_tile_rowmajor(at, S_, 16, 0, lane);
                v8f acc0 = {};
                v8f acc1 = {};
                acc0 = __builtin_amdgcn_wmma_f32_16x16x32_f16(
                    false, a0, false, bm, (short)0, acc0, false, false);
                acc1 = __builtin_amdgcn_wmma_f32_16x16x32_f16(
                    false, a1, false, bm, (short)0, acc1, false, false);
                int n  = h * HD_ + nt * 16 + (lane & 15);
                int mh = (lane & 16) ? 8 : 0;
#pragma unroll
                for (int r = 0; r < 8; ++r) {
                    A16[(size_t)(mh + r) * H_ + n]      = (f16)acc0[r];
                    A16[(size_t)(16 + mh + r) * H_ + n] = (f16)acc1[r];
                }
            }
        }
        __syncthreads();

        // (f) out-proj + residual: xf32 += o @ Wo^T + bo
        gemm_x_wT<true, false>(A16, H_, Wo, H_, bo, nullptr, 0, xf32, H_, 32, wave, lane);
        __syncthreads();

        // (g) LN1 (also stages x -> f16 for FF1)
        layernorm_rows(xf32, A16, ln1_w + (size_t)i * H_, ln1_b + (size_t)i * H_, tid);
        __syncthreads();

        // (h) FF1: hidden = relu(x @ W1^T + b1) -> O16[32 x 1024]
        gemm_x_wT<false, true>(A16, H_, W1, H_, b1, O16, 2 * H_, nullptr, 0, 64, wave, lane);
        __syncthreads();

        // (i) FF2 + residual: xf32 += hidden @ W2^T + b2   (K = 1024)
        gemm_x_wT<true, false>(O16, 2 * H_, W2, 2 * H_, b2, nullptr, 0, xf32, H_, 32, wave, lane);
        __syncthreads();

        // (j) LN2
        layernorm_rows(xf32, nullptr, ln2_w + (size_t)i * H_, ln2_b + (size_t)i * H_, tid);
        __syncthreads();
    }

    // ---- Heads on the last token (row 31) ----
    const int   hidx = (lid & 1) ? (lid - 1) / 2 : 0;
    const float* last = xf32 + (size_t)(S_ - 1) * H_;
    if (tid < E_) {
        const float* wr = head_w + ((size_t)hidx * E_ + tid) * H_;
        float acc = head_b[(size_t)hidx * E_ + tid];
        for (int k = 0; k < H_; ++k) acc += last[k] * wr[k];
        out_pred[(size_t)b * E_ + tid] = acc;
    } else if (tid == E_) {
        float z = conf_b[0];
        for (int k = 0; k < H_; ++k) z += last[k] * conf_w[k];
        out_conf[b] = 1.0f / (1.0f + __expf(-z));
    }
}

// ---------------------------------------------------------------------------
extern "C" void kernel_launch(void* const* d_in, const int* in_sizes, int n_in,
                              void* d_out, int out_size, void* d_ws, size_t ws_size,
                              hipStream_t stream) {
    (void)in_sizes; (void)n_in; (void)out_size; (void)ws_size;

    const int*   expert_sequence = (const int*)  d_in[0];
    const int*   layer_id        = (const int*)  d_in[1];
    const float* expert_table    = (const float*)d_in[2];
    const float* layer_table     = (const float*)d_in[3];
    const float* pos_table       = (const float*)d_in[4];
    const float* qkv_w           = (const float*)d_in[5];
    const float* qkv_b           = (const float*)d_in[6];
    const float* out_w           = (const float*)d_in[7];
    const float* out_b           = (const float*)d_in[8];
    const float* ln1_w           = (const float*)d_in[9];
    const float* ln1_b           = (const float*)d_in[10];
    const float* ff1_w           = (const float*)d_in[11];
    const float* ff1_b           = (const float*)d_in[12];
    const float* ff2_w           = (const float*)d_in[13];
    const float* ff2_b           = (const float*)d_in[14];
    const float* ln2_w           = (const float*)d_in[15];
    const float* ln2_b           = (const float*)d_in[16];
    const float* head_w          = (const float*)d_in[17];
    const float* head_b          = (const float*)d_in[18];
    const float* conf_w          = (const float*)d_in[19];
    const float* conf_b          = (const float*)d_in[20];

    // Workspace layout (f16 weight copies; ~25.2 MB, L2-resident thereafter)
    const size_t nQKV = (size_t)NL_ * 3 * H_ * H_;   // 4,718,592
    const size_t nOUT = (size_t)NL_ * H_ * H_;       // 1,572,864
    const size_t nFF1 = (size_t)NL_ * 2 * H_ * H_;   // 3,145,728
    const size_t nFF2 = (size_t)NL_ * H_ * 2 * H_;   // 3,145,728
    f16* ws   = (f16*)d_ws;
    f16* wqkv = ws;
    f16* wout = wqkv + nQKV;
    f16* wff1 = wout + nOUT;
    f16* wff2 = wff1 + nFF1;

    cvt_f32_f16<<<2048, 256, 0, stream>>>(qkv_w, wqkv, (int)nQKV);
    cvt_f32_f16<<<2048, 256, 0, stream>>>(out_w, wout, (int)nOUT);
    cvt_f32_f16<<<2048, 256, 0, stream>>>(ff1_w, wff1, (int)nFF1);
    cvt_f32_f16<<<2048, 256, 0, stream>>>(ff2_w, wff2, (int)nFF2);

    float* out_pred = (float*)d_out;                 // (B,E)
    float* out_conf = out_pred + (size_t)B_ * E_;    // (B,)

    const size_t smemBytes = 65536 /*xf32*/ + 65536 /*A16*/ + 98304 /*O16*/; // 224 KB
    fused_transformer<<<B_, 256, smemBytes, stream>>>(
        expert_sequence, layer_id, expert_table, layer_table, pos_table,
        wqkv, qkv_b, wout, out_b, ln1_w, ln1_b,
        wff1, ff1_b, wff2, ff2_b, ln2_w, ln2_b,
        head_w, head_b, conf_w, conf_b,
        out_pred, out_conf);
}
